// LSTMModel_58892591563446
// MI455X (gfx1250) — matrix operand order repaired and checked by reading, not compiled, not run on hardware
//
#include <hip/hip_runtime.h>
#include <hip/hip_bf16.h>
#include <cstdint>

// ---------------------------------------------------------------------------
// 2-layer LSTM + cosine head for MI455X (gfx1250, wave32, WMMA f16 16x16x32).
//
// Serial bottleneck = 2048 dependent [64,2048]xK GEMM steps. Weights are
// pre-packed f16 Wcat=[W_ih|W_hh] (7 MB -> L2 resident across all steps).
// Step kernel: 8 blocks x 4 waves; each wave owns one 16-column h-slice,
// all 4 gates, all 4 M-tiles (16 accumulators). B fragments are wave-unique
// (Wcat read exactly once per step chip-wide); the block-uniform A chunk
// (64x32 f16) is staged through double-buffered LDS with async-to-LDS copies
// (GLOBAL_LOAD_ASYNC_TO_LDS_B128 + s_wait_asynccnt), giving coalesced global
// traffic + ds_load_b128 fragment reads.
// ---------------------------------------------------------------------------

typedef __attribute__((ext_vector_type(16))) _Float16 v16h;
typedef __attribute__((ext_vector_type(8)))  float    v8f;
typedef __attribute__((ext_vector_type(4)))  int      v4i;

#define B_   64
#define T_   1024
#define DIN  256
#define H_   512
#define DOUT 256
#define G4   (4 * H_)   // 2048 gate columns

#if defined(__has_builtin)
#if __has_builtin(__builtin_amdgcn_global_load_async_to_lds_b128)
#define HAS_ASYNC_LDS 1
#endif
#endif

#define AS_GLOBAL __attribute__((address_space(1)))
#define AS_LDS    __attribute__((address_space(3)))

union Frag16 {
  v16h  v;
  uint4 q[2];
};

__device__ __forceinline__ float sigmoidf_(float x) {
  return 1.0f / (1.0f + __expf(-x));
}

// Stage one 64x32 f16 A-chunk (rows 0..63, cols k..k+31 of a strided matrix)
// into LDS, 16 B per thread per round (128 threads x 2 rounds = 4 KB).
__device__ __forceinline__ void stage_chunk(const _Float16* __restrict__ aBase,
                                            int strideA, int k,
                                            _Float16* dst, int tid) {
#pragma unroll
  for (int rnd = 0; rnd < 2; ++rnd) {
    const int e   = (tid + rnd * 128) * 8;  // f16 element offset (16B granule)
    const int row = e >> 5;
    const int col = e & 31;
    const _Float16* g = aBase + (size_t)row * strideA + k + col;
#if defined(HAS_ASYNC_LDS)
    __builtin_amdgcn_global_load_async_to_lds_b128(
        (AS_GLOBAL v4i*)(_Float16*)g,     // drop const, cast to global int4*
        (AS_LDS v4i*)(dst + e),           // LDS destination as local int4*
        0, 0);
#else
    *(uint4*)(dst + e) = *(const uint4*)g;
#endif
  }
}

__device__ __forceinline__ void stage_wait_sync() {
#if defined(HAS_ASYNC_LDS)
#if __has_builtin(__builtin_amdgcn_s_wait_asynccnt)
  __builtin_amdgcn_s_wait_asynccnt(0);
#else
  asm volatile("s_wait_asynccnt 0" ::: "memory");
#endif
#endif
  __syncthreads();
}

// --------------------------- prep kernels ----------------------------------

__global__ void f32_to_f16_kernel(const float* __restrict__ in,
                                  _Float16* __restrict__ out, size_t n) {
  for (size_t i = blockIdx.x * (size_t)blockDim.x + threadIdx.x; i < n;
       i += (size_t)gridDim.x * blockDim.x)
    out[i] = (_Float16)in[i];
}

// Wcat[row][0..kx) = W_ih[row], Wcat[row][kx..kx+512) = W_hh[row]  (f32->f16)
__global__ void build_wcat_kernel(const float* __restrict__ wih,
                                  const float* __restrict__ whh,
                                  _Float16* __restrict__ wcat, int kx) {
  const int kw = kx + H_;
  const size_t total = (size_t)G4 * kw;
  for (size_t i = blockIdx.x * (size_t)blockDim.x + threadIdx.x; i < total;
       i += (size_t)gridDim.x * blockDim.x) {
    int row = (int)(i / kw);
    int col = (int)(i % kw);
    float v = (col < kx) ? wih[(size_t)row * kx + col]
                         : whh[(size_t)row * H_ + (col - kx)];
    wcat[i] = (_Float16)v;
  }
}

__global__ void bias_sum_kernel(const float* __restrict__ a,
                                const float* __restrict__ b,
                                float* __restrict__ o, int n) {
  int i = blockIdx.x * blockDim.x + threadIdx.x;
  if (i < n) o[i] = a[i] + b[i];
}

// One wave per row: L2-normalize f32 row -> f16 row.
__global__ void normalize_rows_f32_to_f16(const float* __restrict__ in,
                                          _Float16* __restrict__ out,
                                          int nrows, int ncols) {
  const int lane = threadIdx.x & 31;
  const int row  = blockIdx.x * (blockDim.x >> 5) + (threadIdx.x >> 5);
  if (row >= nrows) return;
  const float* p = in + (size_t)row * ncols;
  float s = 0.0f;
  for (int k = lane; k < ncols; k += 32) { float v = p[k]; s += v * v; }
#pragma unroll
  for (int off = 16; off > 0; off >>= 1) s += __shfl_xor(s, off, 32);
  const float inv = 1.0f / fmaxf(sqrtf(s), 1e-12f);
  _Float16* q = out + (size_t)row * ncols;
  for (int k = lane; k < ncols; k += 32) q[k] = (_Float16)(p[k] * inv);
}

// One wave per row: L2-normalize an f16 row in place (sumsq in f32).
__global__ void normalize_rows_f16_inplace(_Float16* __restrict__ p,
                                           int nrows, int ncols) {
  const int lane = threadIdx.x & 31;
  const int row  = blockIdx.x * (blockDim.x >> 5) + (threadIdx.x >> 5);
  if (row >= nrows) return;
  _Float16* r = p + (size_t)row * ncols;
  float s = 0.0f;
  for (int k = lane; k < ncols; k += 32) { float v = (float)r[k]; s += v * v; }
#pragma unroll
  for (int off = 16; off > 0; off >>= 1) s += __shfl_xor(s, off, 32);
  const float inv = 1.0f / fmaxf(sqrtf(s), 1e-12f);
  for (int k = lane; k < ncols; k += 32)
    r[k] = (_Float16)((float)r[k] * inv);
}

// --------------------------- LSTM step -------------------------------------
// gates = [x_t | h_prev] @ Wcat^T + bias ; i,f,g,o update done in-register.
// grid.x = H/64 = 8; block = 128 threads = 4 waves.
// Wave w owns h-columns [blk*64 + 16w, +16), all 4 gates, all M (4 tiles).
__global__ __launch_bounds__(128)
void lstm_step_kernel(const _Float16* __restrict__ xA, int xRowStride,
                      const _Float16* __restrict__ hPrev,   // [64,512] f16
                      const _Float16* __restrict__ Wcat,    // [2048, kx+512]
                      const float* __restrict__ bias,       // [2048]
                      float* __restrict__ cState,           // [64,512] f32
                      _Float16* __restrict__ hOut,          // [64,512] f16
                      int kx, int hasH) {
  __shared__ __align__(16) _Float16 sA[2][64 * 32];  // double-buffered A chunk

  const int tid     = threadIdx.x;
  const int lane    = tid & 31;
  const int wave    = tid >> 5;
  const int colBase = blockIdx.x * 64 + wave * 16;   // h-column slice
  const int kw      = kx + H_;                       // Wcat row stride
  const int lrow    = lane & 15;
  const int lhi     = lane >> 4;                     // 0 or 1

  v8f acc[4][4] = {};                                // [gate][mtile]

  // B-row base pointers: B column n = Wcat row (gate*512 + colBase + n)
  const _Float16* bRow[4];
#pragma unroll
  for (int g = 0; g < 4; ++g)
    bRow[g] = Wcat + (size_t)(g * H_ + colBase + lrow) * kw + lhi * 8;

  const int nx   = kx >> 5;            // x-chunks of K=32
  const int nh   = hasH ? (H_ >> 5) : 0;
  const int ntot = nx + nh;

  int buf = 0;
  stage_chunk(xA, xRowStride, 0, sA[0], tid);
  stage_wait_sync();

  for (int c = 0; c < ntot; ++c) {
    // prefetch next A chunk into the other buffer (async)
    if (c + 1 < ntot) {
      if (c + 1 < nx)
        stage_chunk(xA, xRowStride, (c + 1) * 32, sA[buf ^ 1], tid);
      else
        stage_chunk(hPrev, H_, (c + 1 - nx) * 32, sA[buf ^ 1], tid);
    }

    // K-offset of this chunk within Wcat rows
    const int wk = (c < nx) ? c * 32 : kx + (c - nx) * 32;

    // A fragments for all 4 M-tiles from LDS (ds_load_b128 x2 each)
    const _Float16* cur = sA[buf];
    Frag16 af[4];
#pragma unroll
    for (int mt = 0; mt < 4; ++mt) {
      const _Float16* ap = cur + (mt * 16 + lrow) * 32 + lhi * 8;
      af[mt].q[0] = *(const uint4*)(ap);
      af[mt].q[1] = *(const uint4*)(ap + 16);
    }

    // one B fragment per gate, reused by all 4 M-tiles
#pragma unroll
    for (int g = 0; g < 4; ++g) {
      Frag16 b;
      const uint4* qb = (const uint4*)(bRow[g] + wk);
      b.q[0] = qb[0];
      b.q[1] = qb[2];
#pragma unroll
      for (int mt = 0; mt < 4; ++mt)
        acc[g][mt] = __builtin_amdgcn_wmma_f32_16x16x32_f16(
            false, af[mt].v, false, b.v, (short)0, acc[g][mt], false, false);
    }

    stage_wait_sync();
    buf ^= 1;
  }

  // ---- gate nonlinearity + cell update, directly on accumulator layout ----
  const int ncol = colBase + lrow;            // D-layout: N = lane & 15
  const float bi = bias[0 * H_ + ncol];
  const float bf = bias[1 * H_ + ncol];
  const float bg = bias[2 * H_ + ncol];
  const float bo = bias[3 * H_ + ncol];
#pragma unroll
  for (int mt = 0; mt < 4; ++mt) {
    const int m0 = mt * 16 + lhi * 8;         // D-layout: M = r (+8 hi half)
#pragma unroll
    for (int r = 0; r < 8; ++r) {
      const int m  = m0 + r;
      const float iv = sigmoidf_(acc[0][mt][r] + bi);
      const float fv = sigmoidf_(acc[1][mt][r] + bf);
      const float gv = tanhf(acc[2][mt][r] + bg);
      const float ov = sigmoidf_(acc[3][mt][r] + bo);
      const float cp = hasH ? cState[(size_t)m * H_ + ncol] : 0.0f;
      const float cn = fv * cp + iv * gv;
      cState[(size_t)m * H_ + ncol] = cn;
      hOut[(size_t)m * H_ + ncol]   = (_Float16)(ov * tanhf(cn));
    }
  }
}

// --------------------------- cosine head GEMM ------------------------------
// out[b*T + t, n] = h2n[t*64 + b, :] . Wfcn[n, :]
// grid = (65536/64, 256/16); 4 waves/block, one 16x16 D tile per wave.
__global__ __launch_bounds__(128)
void head_gemm_kernel(const _Float16* __restrict__ A,   // [65536, 512]
                      const _Float16* __restrict__ Wn,  // [256, 512]
                      float* __restrict__ out) {        // [B*T, 256]
  const int lane  = threadIdx.x & 31;
  const int wave  = threadIdx.x >> 5;
  const int mbase = blockIdx.x * 64 + wave * 16;
  const int n0    = blockIdx.y * 16;
  const int lrow  = lane & 15;
  const int lhi   = lane >> 4;

  const _Float16* aRow = A  + (size_t)(mbase + lrow) * H_ + lhi * 8;
  const _Float16* bRow = Wn + (size_t)(n0 + lrow)    * H_ + lhi * 8;

  v8f acc = {};
  for (int k = 0; k < H_; k += 32) {
    Frag16 a, b;
    a.q[0] = *(const uint4*)(aRow + k);
    a.q[1] = *(const uint4*)(aRow + k + 16);
    const uint4* qb = (const uint4*)(bRow + k);
    b.q[0] = qb[0];
    b.q[1] = qb[2];
    acc = __builtin_amdgcn_wmma_f32_16x16x32_f16(
        false, a.v, false, b.v, (short)0, acc, false, false);
  }

  const int n  = n0 + lrow;
  const int m0 = mbase + lhi * 8;
#pragma unroll
  for (int r = 0; r < 8; ++r) {
    const int m = m0 + r;          // m = t*64 + b  ([T,B,H] layout)
    const int b = m & 63;
    const int t = m >> 6;
    out[((size_t)b * T_ + t) * DOUT + n] = acc[r];
  }
}

// --------------------------- host driver -----------------------------------

extern "C" void kernel_launch(void* const* d_in, const int* in_sizes, int n_in,
                              void* d_out, int out_size, void* d_ws,
                              size_t ws_size, hipStream_t stream) {
  (void)in_sizes; (void)n_in; (void)out_size; (void)ws_size;

  const float* x    = (const float*)d_in[0];
  const float* Wih1 = (const float*)d_in[1];
  const float* Whh1 = (const float*)d_in[2];
  const float* bih1 = (const float*)d_in[3];
  const float* bhh1 = (const float*)d_in[4];
  const float* Wih2 = (const float*)d_in[5];
  const float* Whh2 = (const float*)d_in[6];
  const float* bih2 = (const float*)d_in[7];
  const float* bhh2 = (const float*)d_in[8];
  const float* Wfc  = (const float*)d_in[9];

  char* ws = (char*)d_ws;
  size_t off = 0;
  auto take = [&](size_t bytes) {
    size_t r = off;
    off += (bytes + 255) & ~(size_t)255;
    return r;
  };
  _Float16* wcat1 = (_Float16*)(ws + take((size_t)G4 * (DIN + H_) * 2)); // 3 MB
  _Float16* wcat2 = (_Float16*)(ws + take((size_t)G4 * (H_ + H_) * 2));  // 4 MB
  float*    bias1 = (float*)   (ws + take((size_t)G4 * 4));
  float*    bias2 = (float*)   (ws + take((size_t)G4 * 4));
  _Float16* x16   = (_Float16*)(ws + take((size_t)B_ * T_ * DIN * 2));   // 32 MB
  _Float16* h1    = (_Float16*)(ws + take((size_t)T_ * B_ * H_ * 2));    // 64 MB
  _Float16* h2    = (_Float16*)(ws + take((size_t)T_ * B_ * H_ * 2));    // 64 MB
  _Float16* wfcn  = (_Float16*)(ws + take((size_t)DOUT * H_ * 2));
  float*    c1    = (float*)   (ws + take((size_t)B_ * H_ * 4));
  float*    c2    = (float*)   (ws + take((size_t)B_ * H_ * 4));

  // ---- prep: pack weights/bias to f16 Wcat (L2-resident), convert x ----
  f32_to_f16_kernel<<<4096, 256, 0, stream>>>(x, x16, (size_t)B_ * T_ * DIN);
  build_wcat_kernel<<<4096, 256, 0, stream>>>(Wih1, Whh1, wcat1, DIN);
  build_wcat_kernel<<<4096, 256, 0, stream>>>(Wih2, Whh2, wcat2, H_);
  bias_sum_kernel<<<8, 256, 0, stream>>>(bih1, bhh1, bias1, G4);
  bias_sum_kernel<<<8, 256, 0, stream>>>(bih2, bhh2, bias2, G4);
  normalize_rows_f32_to_f16<<<DOUT / 8, 256, 0, stream>>>(Wfc, wfcn, DOUT, H_);

  // ---- layer 1 recurrence: x layout [B,T,D], h1 layout [T,B,H] ----
  for (int t = 0; t < T_; ++t) {
    lstm_step_kernel<<<H_ / 64, 128, 0, stream>>>(
        x16 + (size_t)t * DIN, T_ * DIN,
        (t > 0) ? (h1 + (size_t)(t - 1) * B_ * H_) : h1,
        wcat1, bias1, c1,
        h1 + (size_t)t * B_ * H_,
        DIN, t > 0 ? 1 : 0);
  }

  // ---- layer 2 recurrence: input rows are h1[t] ----
  for (int t = 0; t < T_; ++t) {
    lstm_step_kernel<<<H_ / 64, 128, 0, stream>>>(
        h1 + (size_t)t * B_ * H_, H_,
        (t > 0) ? (h2 + (size_t)(t - 1) * B_ * H_) : h2,
        wcat2, bias2, c2,
        h2 + (size_t)t * B_ * H_,
        H_, t > 0 ? 1 : 0);
  }

  // ---- head: normalize h2 rows in place, then cosine GEMM ----
  normalize_rows_f16_inplace<<<(T_ * B_) / 8, 256, 0, stream>>>(h2, T_ * B_, H_);
  dim3 hg(T_ * B_ / 64, DOUT / 16);
  head_gemm_kernel<<<hg, 128, 0, stream>>>(h2, wfcn, (float*)d_out);
}
